// MPNEncoder_68977174774322
// MI455X (gfx1250) — matrix operand order repaired and checked by reading, not compiled, not run on hardware
//
#include <hip/hip_runtime.h>
#include <hip/hip_bf16.h>

#define H 256
#define NB 6
#define FA 133
#define FB 147
#define DEPTHM1 3   // DEPTH-1

typedef __attribute__((ext_vector_type(16))) __bf16       v16bf;
typedef __attribute__((ext_vector_type(8)))  float        v8f;
typedef __attribute__((ext_vector_type(4)))  unsigned int uv4;

union Frag32B { uv4 q[2]; v16bf v; };

__device__ __forceinline__ unsigned short f2bf(float x) {
    unsigned int u = __float_as_uint(x);
    u = (u + 0x7FFFu + ((u >> 16) & 1u)) >> 16;   // round-to-nearest-even
    return (unsigned short)u;
}

#if defined(__has_builtin)
#if __has_builtin(__builtin_amdgcn_cvt_pk_bf16_f32)
#define HAVE_CVT_PK_BF16 1
#endif
#endif

__device__ __forceinline__ unsigned int pack2bf(float a, float b) {
#ifdef HAVE_CVT_PK_BF16
    auto r = __builtin_amdgcn_cvt_pk_bf16_f32(a, b);   // v_cvt_pk_bf16_f32
    union { decltype(r) v; unsigned int u; } c;
    c.v = r;
    return c.u;
#else
    return (unsigned int)f2bf(a) | ((unsigned int)f2bf(b) << 16);
#endif
}

// ---------------------------------------------------------------------------
// out[M x 256] = act( A[M x Kpad](f32) @ W(bf16, transposed+padded) + addend + bias )
// A operand: for FULLK, up to three [M x 256] f32 matrices selected by k-chunk
// (k0 >> 8) -> fused "concat" GEMM with Kpad = 256 * n_chunks. For the generic
// path (K=133/147), only A0 is used with zero-padding to Kpad.
// Wt layout: [256 rows = output col][Kpad] bf16, zero-padded for k >= K.
// Block: 256 threads = 8 waves; block tile 128x128; wave tile 64x32 (4x2 WMMA).
// Double-buffered LDS; branch-free A staging (clamped rows feed only masked
// output rows; pad-k zeroed via cndmask).
// ---------------------------------------------------------------------------
template<bool FULLK>
__global__ __launch_bounds__(256)
void wmma_gemm(const float* __restrict__ A0,
               const float* __restrict__ A1,       // k-chunk 1 (fused) or null
               const float* __restrict__ A2,       // k-chunk 2 (fused) or null
               const unsigned short* __restrict__ Wt,
               const float* __restrict__ addend,   // [M x 256] or null
               const float* __restrict__ bias,     // [256] or null
               float* __restrict__ out,            // [M x 256]
               float* __restrict__ out2,           // optional duplicate output
               int Mrows, int K, int Kpad, int relu)
{
    __shared__ __align__(16) unsigned short As[2][128 * 32];   // [row][k]
    __shared__ __align__(16) unsigned short Bs[2][128 * 32];   // [col][k]

    const int tid  = threadIdx.x;
    const int lane = tid & 31;
    const int wave = tid >> 5;      // 0..7
    const int wrow = wave >> 2;     // 0..1 -> 64 rows each
    const int wcol = wave & 3;      // 0..3 -> 32 cols each

    const int rowBlock = blockIdx.x * 128;
    const int colBlock = blockIdx.y * 128;

    // staging thread mapping: 2 threads per row/col, 16 k each
    const int r_st  = tid >> 1;
    const int kk_st = (tid & 1) * 16;
    const int growClamped = min(rowBlock + r_st, Mrows - 1);
    const float* __restrict__ arow0 = A0 + (size_t)growClamped * (size_t)K;
    const float* arow1 = A1 ? A1 + (size_t)growClamped * 256 : nullptr;
    const float* arow2 = A2 ? A2 + (size_t)growClamped * 256 : nullptr;
    const unsigned short* __restrict__ wrowp =
        Wt + (size_t)(colBlock + r_st) * (size_t)Kpad;

    auto stage = [&](int buf, int k0) {
        unsigned int pk[8];
        if (FULLK) {
            const float* base = arow0;
            const int kc = k0 >> 8;                 // 256-wide k-chunks
            if (kc == 1) base = arow1;
            if (kc == 2) base = arow2;
            const float4* src = (const float4*)(base + (k0 & 255) + kk_st);
            #pragma unroll
            for (int i = 0; i < 4; ++i) {
                float4 f = src[i];
                pk[2 * i + 0] = pack2bf(f.x, f.y);
                pk[2 * i + 1] = pack2bf(f.z, f.w);
            }
        } else {
            #pragma unroll
            for (int i = 0; i < 8; ++i) {
                int ka = k0 + kk_st + 2 * i;
                int kb = ka + 1;
                float va = arow0[min(ka, K - 1)];
                float vb = arow0[min(kb, K - 1)];
                va = (ka < K) ? va : 0.0f;          // cndmask, no branch
                vb = (kb < K) ? vb : 0.0f;
                pk[i] = pack2bf(va, vb);
            }
        }
        uv4* da = (uv4*)(&As[buf][r_st * 32 + kk_st]);
        uv4 lo = {pk[0], pk[1], pk[2], pk[3]};
        uv4 hi = {pk[4], pk[5], pk[6], pk[7]};
        da[0] = lo;
        da[1] = hi;
        const uv4* sb = (const uv4*)(wrowp + k0 + kk_st);
        uv4* db = (uv4*)(&Bs[buf][r_st * 32 + kk_st]);
        db[0] = sb[0];
        db[1] = sb[1];
    };

    v8f zf = {};
    v8f acc[4][2];
    #pragma unroll
    for (int t = 0; t < 4; ++t)
        #pragma unroll
        for (int u = 0; u < 2; ++u) acc[t][u] = zf;

    const int lr  = lane & 15;
    const int khA = (lane & 16) ? 8 : 0;    // A frag: lanes 16-31 hold K 8..15,24..31
    const int khB = (lane & 16) ? 8 : 0;    // B frag: lanes 16-31 hold K 8..15,24..31

    const int nk = Kpad >> 5;
    stage(0, 0);
    __syncthreads();

    for (int it = 0; it < nk; ++it) {
        const int cur = it & 1;
        const unsigned short* AsB = As[cur];
        const unsigned short* BsB = Bs[cur];

        // fragment loads from current buffer first...
        Frag32B afr[4], bfr[2];
        #pragma unroll
        for (int t = 0; t < 4; ++t) {
            const unsigned short* p = AsB + (wrow * 64 + t * 16 + lr) * 32 + khA;
            afr[t].q[0] = *(const uv4*)(p);        // K khA..khA+7
            afr[t].q[1] = *(const uv4*)(p + 16);   // K khA+16..khA+23
        }
        #pragma unroll
        for (int u = 0; u < 2; ++u) {
            const unsigned short* p = BsB + (wcol * 32 + u * 16 + lr) * 32 + khB;
            bfr[u].q[0] = *(const uv4*)(p);        // K khB..khB+7
            bfr[u].q[1] = *(const uv4*)(p + 16);   // K khB+16..khB+23
        }

        // ...then stage next tile into the other buffer (overlaps with WMMA)
        if (it + 1 < nk) stage(cur ^ 1, (it + 1) << 5);

        #pragma unroll
        for (int t = 0; t < 4; ++t)
            #pragma unroll
            for (int u = 0; u < 2; ++u)
                acc[t][u] = __builtin_amdgcn_wmma_f32_16x16x32_bf16(
                    false, afr[t].v, false, bfr[u].v, (short)0, acc[t][u],
                    false, false);

        __syncthreads();
    }

    const int rAdd = (lane & 16) ? 8 : 0;
    #pragma unroll
    for (int t = 0; t < 4; ++t) {
        #pragma unroll
        for (int u = 0; u < 2; ++u) {
            int col = colBlock + wcol * 32 + u * 16 + lr;
            #pragma unroll
            for (int r = 0; r < 8; ++r) {
                int row = rowBlock + wrow * 64 + t * 16 + r + rAdd;
                if (row < Mrows) {
                    float v = acc[t][u][r];
                    size_t idx = (size_t)row * H + col;
                    if (addend) v += addend[idx];
                    if (bias)   v += bias[col];
                    if (relu)   v = fmaxf(v, 0.0f);
                    out[idx] = v;
                    if (out2) out2[idx] = v;
                }
            }
        }
    }
}

// W[K x N] f32 -> Wt[N x Kpad] bf16 (transposed, zero-padded), W row offset for slices
__global__ void conv_w_t(const float* __restrict__ W, unsigned short* __restrict__ Wt,
                         int K, int N, int Kpad, int rowOff)
{
    int t = blockIdx.x * blockDim.x + threadIdx.x;
    if (t >= N * Kpad) return;
    int n = t / Kpad, k = t % Kpad;
    float v = (k < K) ? W[(size_t)(rowOff + k) * N + n] : 0.0f;
    Wt[(size_t)n * Kpad + k] = f2bf(v);
}

// out[a][h] = (base ? base : 0) + sum_j(msg_bond[a2b[a][j]][h]) * max_j(...)
__global__ __launch_bounds__(256)
void agg_sum_max(const float* __restrict__ msg_bond, const int* __restrict__ a2b,
                 const float* __restrict__ base, float* __restrict__ out)
{
    int a = blockIdx.x;
    int h = threadIdx.x;
    int b0 = a2b[a * NB];
    float v = msg_bond[(size_t)b0 * H + h];
    float s = v, m = v;
    #pragma unroll
    for (int j = 1; j < NB; ++j) {
        int b = a2b[a * NB + j];
        float w = msg_bond[(size_t)b * H + h];
        s += w;
        m = fmaxf(m, w);
    }
    float r = s * m;
    if (base) r += base[(size_t)a * H + h];
    out[(size_t)a * H + h] = r;
}

// tmp[b][h] = msg_atom[b2a[b]][h] - msg_bond[b2revb[b]][h]
__global__ __launch_bounds__(256)
void bond_gather(const float* __restrict__ msg_atom, const float* __restrict__ msg_bond,
                 const int* __restrict__ b2a, const int* __restrict__ b2revb,
                 float* __restrict__ outp)
{
    int b = blockIdx.x, h = threadIdx.x;
    int a  = b2a[b];
    int rb = b2revb[b];
    outp[(size_t)b * H + h] =
        msg_atom[(size_t)a * H + h] - msg_bond[(size_t)rb * H + h];
}

__global__ void prefix_sizes(const int* __restrict__ sizes, int* __restrict__ offs, int M)
{
    if (blockIdx.x == 0 && threadIdx.x == 0) {
        int acc = 0;
        for (int i = 0; i < M; ++i) { offs[i] = acc; acc += sizes[i]; }
        offs[M] = acc;
    }
}

__global__ __launch_bounds__(256)
void pool_mean(const float* __restrict__ hid, const int* __restrict__ offs,
               float* __restrict__ out)
{
    int m = blockIdx.x, h = threadIdx.x;
    int s0 = offs[m], s1 = offs[m + 1];
    float s = 0.0f;
    for (int a = s0; a < s1; ++a) s += hid[(size_t)a * H + h];
    out[(size_t)m * H + h] = s / (float)(s1 - s0);
}

extern "C" void kernel_launch(void* const* d_in, const int* in_sizes, int n_in,
                              void* d_out, int out_size, void* d_ws, size_t ws_size,
                              hipStream_t stream)
{
    const float* f_atoms = (const float*)d_in[0];
    const float* f_bonds = (const float*)d_in[1];
    const float* W_ia    = (const float*)d_in[2];
    const float* W_ib    = (const float*)d_in[3];
    const float* W_h     = (const float*)d_in[4];
    const float* W_o     = (const float*)d_in[5];
    const float* b_o     = (const float*)d_in[6];
    const float* W_lr    = (const float*)d_in[7];
    const int* a2b       = (const int*)d_in[8];
    const int* b2a       = (const int*)d_in[9];
    const int* b2revb    = (const int*)d_in[10];
    const int* sizes     = (const int*)d_in[11];

    const int Arows = in_sizes[0] / FA;
    const int Brows = in_sizes[1] / FB;
    const int Mmol  = in_sizes[11];

    // ---- carve workspace (256B aligned chunks)
    char* p = (char*)d_ws;
    auto carve = [&](size_t bytes) {
        char* r = p;
        p += (bytes + 255) & ~((size_t)255);
        return (void*)r;
    };
    int*            offs  = (int*)carve((size_t)(Mmol + 1) * 4);
    unsigned short* wia_t = (unsigned short*)carve((size_t)256 * 160 * 2);
    unsigned short* wib_t = (unsigned short*)carve((size_t)256 * 160 * 2);
    unsigned short* wh_t  = (unsigned short*)carve((size_t)DEPTHM1 * 256 * 256 * 2);
    unsigned short* wo_t  = (unsigned short*)carve((size_t)256 * 256 * 2);
    unsigned short* wlr_t = (unsigned short*)carve((size_t)256 * 768 * 2);  // [col][768]
    const size_t bytesA = (size_t)Arows * H * sizeof(float);
    const size_t bytesB = (size_t)Brows * H * sizeof(float);
    float* input_atom   = (float*)carve(bytesA);
    float* message_atom = (float*)carve(bytesA);
    float* input_bond   = (float*)carve(bytesB);
    float* message_bond = (float*)carve(bytesB);
    float* bond_tmp     = (float*)carve(bytesB);
    float* aggbuf       = (float*)carve(bytesA);
    float* out1         = (float*)carve(bytesA);
    float* hid          = bond_tmp;   // reuse: bond_tmp dead after depth loop

    // ---- pre-convert weights to transposed/padded bf16
    {
        int thr = 256;
        conv_w_t<<<(256 * 160 + thr - 1) / thr, thr, 0, stream>>>(W_ia, wia_t, FA, H, 160, 0);
        conv_w_t<<<(256 * 160 + thr - 1) / thr, thr, 0, stream>>>(W_ib, wib_t, FB, H, 160, 0);
        for (int d = 0; d < DEPTHM1; ++d)
            conv_w_t<<<(256 * 256 + thr - 1) / thr, thr, 0, stream>>>(
                W_h, wh_t + (size_t)d * 256 * 256, 256, H, 256, d * 256);
        conv_w_t<<<(256 * 256 + thr - 1) / thr, thr, 0, stream>>>(W_o, wo_t, 256, H, 256, 0);
        conv_w_t<<<(256 * 768 + thr - 1) / thr, thr, 0, stream>>>(W_lr, wlr_t, 768, H, 768, 0);
    }

    dim3 blk(256);
    dim3 gA((Arows + 127) / 128, 2);
    dim3 gB((Brows + 127) / 128, 2);

    // input_atom = relu(f_atoms @ W_ia); message_atom = copy (fused via out2)
    wmma_gemm<false><<<gA, blk, 0, stream>>>(f_atoms, nullptr, nullptr, wia_t,
                                             nullptr, nullptr,
                                             input_atom, message_atom,
                                             Arows, FA, 160, 1);
    // input_bond = relu(f_bonds @ W_ib); message_bond = copy (fused via out2)
    wmma_gemm<false><<<gB, blk, 0, stream>>>(f_bonds, nullptr, nullptr, wib_t,
                                             nullptr, nullptr,
                                             input_bond, message_bond,
                                             Brows, FB, 160, 1);

    // message-passing depth loop
    for (int d = 0; d < DEPTHM1; ++d) {
        agg_sum_max<<<Arows, blk, 0, stream>>>(message_bond, a2b, message_atom, message_atom);
        bond_gather<<<Brows, blk, 0, stream>>>(message_atom, message_bond, b2a, b2revb, bond_tmp);
        wmma_gemm<true><<<gB, blk, 0, stream>>>(bond_tmp, nullptr, nullptr,
                                                wh_t + (size_t)d * 256 * 256,
                                                input_bond, nullptr,
                                                message_bond, nullptr,
                                                Brows, 256, 256, 1);
    }

    // readout: fused concat GEMM: out1 = [agg | message_atom | input_atom] @ W_lr
    agg_sum_max<<<Arows, blk, 0, stream>>>(message_bond, a2b, nullptr, aggbuf);
    wmma_gemm<true><<<gA, blk, 0, stream>>>(aggbuf, message_atom, input_atom, wlr_t,
                                            nullptr, nullptr,
                                            out1, nullptr,
                                            Arows, 256, 768, 0);

    // atom_hiddens = relu(out1 @ W_o + b_o)
    wmma_gemm<true><<<gA, blk, 0, stream>>>(out1, nullptr, nullptr, wo_t,
                                            nullptr, b_o,
                                            hid, nullptr,
                                            Arows, 256, 256, 1);

    // ragged per-molecule mean pooling
    prefix_sizes<<<1, 1, 0, stream>>>(sizes, offs, Mmol);
    pool_mean<<<Mmol, blk, 0, stream>>>(hid, offs, (float*)d_out);

    (void)n_in; (void)out_size; (void)ws_size;
}
